// environment_network_84378927497725
// MI455X (gfx1250) — matrix operand order repaired and checked by reading, not compiled, not run on hardware
//
#include <hip/hip_runtime.h>

// ---------------------------------------------------------------------------
// Hypergraph message-passing network for MI455X (gfx1250, wave32, WMMA).
//   send    = a0 + a2 ; receive = a0 + a1
//   m   = relu((x*send) @ W1^T + b1)            <- bf16 WMMA GEMM
//   e   = mean over pairs of m[v_ids] by e_ids  <- f32 atomics (L2-resident)
//   mi  = mean over pairs of e[e_ids] by v_ids  <- f32 atomics (L2-resident)
//   h   = relu(x @ Wu^T + bu + mi*receive)      <- bf16 WMMA GEMM, fused epilogue
//
// B matrices are pre-packed into per-lane WMMA register layout so GEMM B loads
// are contiguous b128s (no strided u16 gathers, no v_mov_b16 packing).
// Each wave computes MT=5 M-tiles against one N-tile, reusing B registers.
// ---------------------------------------------------------------------------

#define NV 100000
#define NE 20000
#define NP 800000
#define FD 128
#define MT 5            // M-tiles per wave (6250 / 5 = 1250 blocks)

typedef __attribute__((ext_vector_type(16))) __bf16 v16bf;
typedef __attribute__((ext_vector_type(8)))  float  v8f;

// ---------------------------------------------------------------------------
// Prep: xs_bf16[v][f] = bf16(x[v][f] * (a0+a2))
// ---------------------------------------------------------------------------
__global__ __launch_bounds__(256) void prep_xs_kernel(
    const float* __restrict__ x, const float* __restrict__ action,
    __bf16* __restrict__ xs)
{
    int tid = blockIdx.x * blockDim.x + threadIdx.x;
    if (tid >= NV * FD) return;
    int v = tid >> 7;                       // / FD
    float send = action[v * 3 + 0] + action[v * 3 + 2];
    xs[tid] = (__bf16)(x[tid] * send);
}

// ---------------------------------------------------------------------------
// Prep: pack W (f32, [j][k]) into per-lane WMMA B-operand layout (bf16).
// Packed index: (((ntile*4 + kstep)*32 + lane)*16 + e)
//   column N = ntile*16 + lane%16
//   K        = kstep*32 + (e/8)*16 + (lane/16)*8 + (e%8)
//   value    = W[N][K]        (B[k][n] = W^T[k][n] = W[n][k])
// Each lane's 16 bf16 values are contiguous & 32B-aligned -> b128 loads.
// ---------------------------------------------------------------------------
__global__ __launch_bounds__(256) void pack_w_kernel(
    const float* __restrict__ W1, const float* __restrict__ Wu,
    __bf16* __restrict__ W1p, __bf16* __restrict__ Wup)
{
    int tid = blockIdx.x * blockDim.x + threadIdx.x;
    if (tid >= FD * FD) return;             // 16384 packed elements
    int e    = tid & 15;
    int lane = (tid >> 4) & 31;
    int ks   = (tid >> 9) & 3;
    int nt   = tid >> 11;                   // 0..7
    int col  = nt * 16 + (lane & 15);
    int k    = ks * 32 + (e >> 3) * 16 + (lane >> 4) * 8 + (e & 7);
    W1p[tid] = (__bf16)W1[col * FD + k];
    Wup[tid] = (__bf16)Wu[col * FD + k];
}

// ---------------------------------------------------------------------------
// A-tile load (16x32 bf16, ISA per-lane layout):
//   lane holds row M = lane%16; element e covers
//   K = (e/8)*16 + (lane/16)*8 + (e%8)  -> two contiguous 16B chunks (b128s)
// ---------------------------------------------------------------------------
__device__ __forceinline__ v16bf load_a_tile(const __bf16* __restrict__ base,
                                             int row, int kb, int half)
{
    v16bf a;
    const __bf16* ap = base + row * FD + kb + half * 8;
#pragma unroll
    for (int e = 0; e < 8; ++e) {
        a[e]     = ap[e];        // K 0..15 half
        a[e + 8] = ap[16 + e];   // K 16..31 half
    }
    return a;
}

// ---------------------------------------------------------------------------
// GEMM1: m = relu(xs @ W1^T + b1)
// grid.x = NV/(16*MT); 8 waves/block; wave w owns N-tile w.
// B registers (4 K-steps) loaded once, reused across MT M-tiles.
// ---------------------------------------------------------------------------
__global__ __launch_bounds__(256) void gemm1_kernel(
    const __bf16* __restrict__ xs, const __bf16* __restrict__ W1p,
    const float* __restrict__ b1, float* __restrict__ m)
{
    const int lane   = threadIdx.x & 31;
    const int ntile  = threadIdx.x >> 5;    // 0..7
    const int mtile0 = blockIdx.x * MT;
    const int half   = lane >> 4;
    const int col    = ntile * 16 + (lane & 15);

    // B operands for all 4 K-steps: contiguous 32B per lane.
    v16bf b[4];
#pragma unroll
    for (int ks = 0; ks < 4; ++ks)
        b[ks] = *(const v16bf*)(W1p + (((ntile * 4 + ks) * 32 + lane) << 4));

    const float bias = b1[col];

#pragma unroll
    for (int t = 0; t < MT; ++t) {
        const int mtile = mtile0 + t;
        const int arow  = mtile * 16 + (lane & 15);
        v8f c = {};
#pragma unroll
        for (int ks = 0; ks < 4; ++ks) {
            v16bf a = load_a_tile(xs, arow, ks * 32, half);
            c = __builtin_amdgcn_wmma_f32_16x16x32_bf16(
                    false, a, false, b[ks], (short)0, c, false, false);
        }
        const int orow = mtile * 16 + half * 8;
#pragma unroll
        for (int r = 0; r < 8; ++r) {
            float v = c[r] + bias;
            m[(orow + r) * FD + col] = v > 0.f ? v : 0.f;
        }
    }
}

// ---------------------------------------------------------------------------
// Pair scatter #1: e_sum[e] += m[v]; counts for both sides (once per pair).
// One wave per pair; each lane handles 4 features (float4 gather).
// ---------------------------------------------------------------------------
__global__ __launch_bounds__(256) void edge_accum_kernel(
    const float* __restrict__ m, const int* __restrict__ v_ids,
    const int* __restrict__ e_ids, float* __restrict__ e_sum,
    float* __restrict__ e_cnt, float* __restrict__ v_cnt)
{
    int gid  = blockIdx.x * blockDim.x + threadIdx.x;
    int p    = gid >> 5;
    int lane = gid & 31;
    if (p >= NP) return;
    int v = v_ids[p];
    int e = e_ids[p];
    const float4 val = ((const float4*)(m + (size_t)v * FD))[lane];
    float* dst = e_sum + (size_t)e * FD + lane * 4;
    atomicAdd(dst + 0, val.x);
    atomicAdd(dst + 1, val.y);
    atomicAdd(dst + 2, val.z);
    atomicAdd(dst + 3, val.w);
    if (lane == 0) {
        atomicAdd(e_cnt + e, 1.0f);
        atomicAdd(v_cnt + v, 1.0f);
    }
}

// ---------------------------------------------------------------------------
// e_mean = e_sum / max(e_cnt, 1)
// ---------------------------------------------------------------------------
__global__ __launch_bounds__(256) void edge_mean_kernel(
    const float* __restrict__ e_sum, const float* __restrict__ e_cnt,
    float* __restrict__ e_mean)
{
    int tid = blockIdx.x * blockDim.x + threadIdx.x;
    if (tid >= NE * FD) return;
    int e = tid >> 7;
    e_mean[tid] = e_sum[tid] / fmaxf(e_cnt[e], 1.0f);
}

// ---------------------------------------------------------------------------
// Pair scatter #2: v_sum[v] += e_mean[e]
// ---------------------------------------------------------------------------
__global__ __launch_bounds__(256) void vert_accum_kernel(
    const float* __restrict__ e_mean, const int* __restrict__ v_ids,
    const int* __restrict__ e_ids, float* __restrict__ v_sum)
{
    int gid  = blockIdx.x * blockDim.x + threadIdx.x;
    int p    = gid >> 5;
    int lane = gid & 31;
    if (p >= NP) return;
    int v = v_ids[p];
    int e = e_ids[p];
    const float4 val = ((const float4*)(e_mean + (size_t)e * FD))[lane];
    float* dst = v_sum + (size_t)v * FD + lane * 4;
    atomicAdd(dst + 0, val.x);
    atomicAdd(dst + 1, val.y);
    atomicAdd(dst + 2, val.z);
    atomicAdd(dst + 3, val.w);
}

// ---------------------------------------------------------------------------
// GEMM2 + fused epilogue:
//   h = relu(x @ Wu^T + bu + (v_sum/max(v_cnt,1)) * receive)
// A loaded from f32 x, converted to bf16 on the fly (8-wide contiguous).
// ---------------------------------------------------------------------------
__global__ __launch_bounds__(256) void gemm2_kernel(
    const float* __restrict__ x, const __bf16* __restrict__ Wup,
    const float* __restrict__ bu, const float* __restrict__ action,
    const float* __restrict__ v_sum, const float* __restrict__ v_cnt,
    float* __restrict__ h)
{
    const int lane   = threadIdx.x & 31;
    const int ntile  = threadIdx.x >> 5;
    const int mtile0 = blockIdx.x * MT;
    const int half   = lane >> 4;
    const int col    = ntile * 16 + (lane & 15);

    v16bf b[4];
#pragma unroll
    for (int ks = 0; ks < 4; ++ks)
        b[ks] = *(const v16bf*)(Wup + (((ntile * 4 + ks) * 32 + lane) << 4));

    const float bias = bu[col];

#pragma unroll
    for (int t = 0; t < MT; ++t) {
        const int mtile = mtile0 + t;
        const int arow  = mtile * 16 + (lane & 15);
        v8f c = {};
#pragma unroll
        for (int ks = 0; ks < 4; ++ks) {
            v16bf a;
            const float* ap = x + (size_t)arow * FD + ks * 32 + half * 8;
#pragma unroll
            for (int e = 0; e < 8; ++e) {
                a[e]     = (__bf16)ap[e];
                a[e + 8] = (__bf16)ap[16 + e];
            }
            c = __builtin_amdgcn_wmma_f32_16x16x32_bf16(
                    false, a, false, b[ks], (short)0, c, false, false);
        }
        const int orow = mtile * 16 + half * 8;
#pragma unroll
        for (int r = 0; r < 8; ++r) {
            int row = orow + r;
            float receive = action[row * 3 + 0] + action[row * 3 + 1];
            float mi = (v_sum[(size_t)row * FD + col] / fmaxf(v_cnt[row], 1.0f)) * receive;
            float v = c[r] + bias + mi;
            h[(size_t)row * FD + col] = v > 0.f ? v : 0.f;
        }
    }
}

// ---------------------------------------------------------------------------
// Launcher
// ---------------------------------------------------------------------------
extern "C" void kernel_launch(void* const* d_in, const int* in_sizes, int n_in,
                              void* d_out, int out_size, void* d_ws, size_t ws_size,
                              hipStream_t stream) {
    (void)in_sizes; (void)n_in; (void)out_size; (void)ws_size;

    const float* x      = (const float*)d_in[0];
    const float* action = (const float*)d_in[1];
    const int*   v_ids  = (const int*)  d_in[2];
    const int*   e_ids  = (const int*)  d_in[3];
    const float* W1     = (const float*)d_in[4];
    const float* b1     = (const float*)d_in[5];
    const float* Wu     = (const float*)d_in[6];
    const float* bu     = (const float*)d_in[7];
    float* h = (float*)d_out;

    // Workspace carve-up (256B aligned)
    char* base = (char*)d_ws;
    size_t off = 0;
    auto carve = [&](size_t bytes) -> char* {
        char* p = base + off;
        off = (off + bytes + 255) & ~(size_t)255;
        return p;
    };
    __bf16* xs    = (__bf16*)carve((size_t)NV * FD * sizeof(__bf16));
    __bf16* W1p   = (__bf16*)carve((size_t)FD * FD * sizeof(__bf16));
    __bf16* Wup   = (__bf16*)carve((size_t)FD * FD * sizeof(__bf16));
    float*  m     = (float*) carve((size_t)NV * FD * sizeof(float));
    float*  e_sum = (float*) carve((size_t)NE * FD * sizeof(float));
    float*  e_mn  = (float*) carve((size_t)NE * FD * sizeof(float));
    float*  e_cnt = (float*) carve((size_t)NE * sizeof(float));
    float*  v_sum = (float*) carve((size_t)NV * FD * sizeof(float));
    float*  v_cnt = (float*) carve((size_t)NV * sizeof(float));

    // Zero the atomic accumulators every call (graph-capture safe).
    hipMemsetAsync(e_sum, 0, (size_t)NE * FD * sizeof(float), stream);
    hipMemsetAsync(e_cnt, 0, (size_t)NE * sizeof(float), stream);
    hipMemsetAsync(v_sum, 0, (size_t)NV * FD * sizeof(float), stream);
    hipMemsetAsync(v_cnt, 0, (size_t)NV * sizeof(float), stream);

    // 1) prep: xs = bf16(x * send); packed bf16 weights in B-operand layout
    prep_xs_kernel<<<(NV * FD + 255) / 256, 256, 0, stream>>>(x, action, xs);
    pack_w_kernel<<<(FD * FD + 255) / 256, 256, 0, stream>>>(W1, Wu, W1p, Wup);

    // 2) GEMM1 (WMMA bf16): m = relu(xs @ W1^T + b1)
    gemm1_kernel<<<NV / (16 * MT), 256, 0, stream>>>(xs, W1p, b1, m);

    // 3) vertex -> hyperedge scatter-sum + counts (1 wave / pair)
    edge_accum_kernel<<<(NP * 32) / 256, 256, 0, stream>>>(
        m, v_ids, e_ids, e_sum, e_cnt, v_cnt);

    // 4) hyperedge mean
    edge_mean_kernel<<<(NE * FD + 255) / 256, 256, 0, stream>>>(e_sum, e_cnt, e_mn);

    // 5) hyperedge -> vertex scatter-sum
    vert_accum_kernel<<<(NP * 32) / 256, 256, 0, stream>>>(e_mn, v_ids, e_ids, v_sum);

    // 6) GEMM2 (WMMA bf16) + fused mean/receive/bias/ReLU epilogue
    gemm2_kernel<<<NV / (16 * MT), 256, 0, stream>>>(x, Wup, bu, action, v_sum, v_cnt, h);
}
